// CausalSelfAttention_48026324303934
// MI455X (gfx1250) — matrix operand order repaired and checked by reading, not compiled, not run on hardware
//
#include <hip/hip_runtime.h>
#include <hip/hip_bf16.h>
#include <stdint.h>

// CDNA5 / gfx1250 wave32 WMMA implementation of causal self-attention fwd.
// All heavy math runs on v_wmma_f32_16x16x32_bf16 (fp32 accumulate).

typedef __attribute__((ext_vector_type(16))) __bf16 v16bf;
typedef __attribute__((ext_vector_type(8)))  float  v8f;

#define DEV __device__ __forceinline__

constexpr int kB   = 2;
constexpr int kT   = 2048;
constexpr int kC   = 1024;
constexpr int kH   = 16;
constexpr int kDh  = 64;
constexpr int kBT  = kB * kT;      // 4096 rows
constexpr int kNq  = 3 * kC;       // 3072 qkv cols

// ---- workspace layout (bytes) ----
constexpr size_t SZ_XB    = (size_t)kBT * kC * 2;       // x bf16
constexpr size_t SZ_WQKVT = (size_t)kC * kNq * 2;       // W_qkv^T bf16
constexpr size_t SZ_WOUTT = (size_t)kC * kC * 2;        // W_out^T bf16
constexpr size_t SZ_QB    = (size_t)kBT * kC * 2;       // q bf16 [B,H,T,Dh]
constexpr size_t SZ_KB    = SZ_QB;                      // k bf16 [B,H,T,Dh]
constexpr size_t SZ_VTB   = SZ_QB;                      // v bf16 [B,H,Dh,T]
constexpr size_t OFF_WQKVT = SZ_XB;
constexpr size_t OFF_WOUTT = OFF_WQKVT + SZ_WQKVT;
constexpr size_t OFF_QB    = OFF_WOUTT + SZ_WOUTT;
constexpr size_t OFF_KB    = OFF_QB + SZ_QB;
constexpr size_t OFF_VTB   = OFF_KB + SZ_KB;
constexpr size_t OFF_YATT  = OFF_VTB + SZ_VTB;

DEV v8f wmma_bf16(v16bf a, v16bf b, v8f c) {
  return __builtin_amdgcn_wmma_f32_16x16x32_bf16(false, a, false, b, (short)0, c,
                                                 false, false);
}

// A-matrix fragment (16x32 bf16): lane = row (lane&15); per half-wave h=lane>>4,
// elems 0..15 hold K = k0+16h .. k0+16h+15 contiguous.
DEV v16bf load_a_frag(const __bf16* base, int row0, int ld, int k0) {
  int lane = threadIdx.x & 31;
  int m = lane & 15, h = lane >> 4;
  const __bf16* p = base + (size_t)(row0 + m) * ld + k0 + h * 16;
  union { v16bf v; uint4 u[2]; } f;
  f.u[0] = *(const uint4*)(p);
  f.u[1] = *(const uint4*)(p + 8);
  return f.v;
}

// B-matrix fragment (32x16 bf16), loaded from B^T stored row-major [N,K]:
// lane = column n (lane&15); half-wave h holds K = k0+16h .. k0+16h+15 contiguous.
DEV v16bf load_b_frag(const __bf16* baseT, int n0, int ld, int k0) {
  int lane = threadIdx.x & 31;
  int n = lane & 15, h = lane >> 4;
  const __bf16* p = baseT + (size_t)(n0 + n) * ld + k0 + h * 16;
  union { v16bf v; uint4 u[2]; } f;
  f.u[0] = *(const uint4*)(p);
  f.u[1] = *(const uint4*)(p + 8);
  return f.v;
}

// ---------------- conversion kernels ----------------
__global__ void cvt_f32_bf16(const float* __restrict__ in, __bf16* __restrict__ out, int n) {
  int i = blockIdx.x * blockDim.x + threadIdx.x;
  if (i < n) out[i] = (__bf16)in[i];
}

// in: [K,N] row-major fp32 -> out: [N,K] row-major bf16
__global__ void cvt_transpose(const float* __restrict__ in, __bf16* __restrict__ out,
                              int K, int N) {
  int i = blockIdx.x * blockDim.x + threadIdx.x;
  if (i < K * N) {
    int k = i / N, n = i - k * N;
    out[(size_t)n * K + k] = (__bf16)in[i];
  }
}

// ---------------- QKV projection GEMM ----------------
// xb [4096,1024] bf16 @ W_qkv -> qkv [4096,3072]; scatter epilogue.
__global__ void qkv_gemm(const __bf16* __restrict__ xb, const __bf16* __restrict__ wT,
                         const float* __restrict__ bias,
                         float* __restrict__ k_out, float* __restrict__ v_out,
                         __bf16* __restrict__ qb, __bf16* __restrict__ kb,
                         __bf16* __restrict__ vTb) {
  int wave = threadIdx.x >> 5;
  int m0 = blockIdx.x * 32;
  int n0 = (blockIdx.y * 4 + wave) * 32;
  v8f c00 = {}, c01 = {}, c10 = {}, c11 = {};
  for (int k0 = 0; k0 < kC; k0 += 32) {
    v16bf a0 = load_a_frag(xb, m0,      kC, k0);
    v16bf a1 = load_a_frag(xb, m0 + 16, kC, k0);
    v16bf b0 = load_b_frag(wT, n0,      kC, k0);
    v16bf b1 = load_b_frag(wT, n0 + 16, kC, k0);
    c00 = wmma_bf16(a0, b0, c00);
    c01 = wmma_bf16(a0, b1, c01);
    c10 = wmma_bf16(a1, b0, c10);
    c11 = wmma_bf16(a1, b1, c11);
  }
  int lane = threadIdx.x & 31;
  int colL = lane & 15, hh = lane >> 4;
  v8f acc[2][2] = {{c00, c01}, {c10, c11}};
  #pragma unroll
  for (int mf = 0; mf < 2; ++mf)
    #pragma unroll
    for (int nf = 0; nf < 2; ++nf)
      #pragma unroll
      for (int r = 0; r < 8; ++r) {
        int row = m0 + 16 * mf + 8 * hh + r;   // [0,4096): b*T+t
        int n   = n0 + 16 * nf + colL;         // [0,3072)
        float val = acc[mf][nf][r] + bias[n];
        int which = n >> 10, cc = n & 1023;
        int h = cc >> 6, d = cc & 63;
        int b = row >> 11, t = row & 2047;
        size_t idx = (((size_t)b * kH + h) * kT + t) * kDh + d;
        __bf16 vb = (__bf16)val;
        if (which == 0) {
          qb[idx] = vb;
        } else if (which == 1) {
          k_out[idx] = val; kb[idx] = vb;
        } else {
          v_out[idx] = val;
          vTb[(((size_t)b * kH + h) * kDh + d) * kT + t] = vb;  // transposed for PV
        }
      }
}

// ---------------- flash attention ----------------
// One wave per 32-row q tile per (b,h). Online softmax, WMMA for QK^T and PV.
__global__ void attn_kernel(const __bf16* __restrict__ qb, const __bf16* __restrict__ kb,
                            const __bf16* __restrict__ vTb, __bf16* __restrict__ yatt) {
  __shared__ __attribute__((aligned(16))) __bf16 plds[4][32 * 32];
  int wave = threadIdx.x >> 5;
  int lane = threadIdx.x & 31;
  int gw = blockIdx.x * 4 + wave;
  int qt = gw & 63;          // T/32 = 64 q tiles
  int bh = gw >> 6;          // [0, 32)
  int q0 = qt * 32;
  const __bf16* qp = qb  + (size_t)bh * kT * kDh;
  const __bf16* kp = kb  + (size_t)bh * kT * kDh;
  const __bf16* vp = vTb + (size_t)bh * kDh * kT;
  int colL = lane & 15, hh = lane >> 4;

  v16bf qa[2][2];
  #pragma unroll
  for (int mf = 0; mf < 2; ++mf)
    #pragma unroll
    for (int ks = 0; ks < 2; ++ks)
      qa[mf][ks] = load_a_frag(qp, q0 + 16 * mf, kDh, 32 * ks);

  v8f o[2][4];
  #pragma unroll
  for (int mf = 0; mf < 2; ++mf)
    #pragma unroll
    for (int nf = 0; nf < 4; ++nf) o[mf][nf] = (v8f){};
  float mrow[2][8], lrow[2][8];
  #pragma unroll
  for (int mf = 0; mf < 2; ++mf)
    #pragma unroll
    for (int r = 0; r < 8; ++r) { mrow[mf][r] = -1.0e30f; lrow[mf][r] = 0.0f; }

  for (int kt = 0; kt <= qt; ++kt) {
    int k0 = kt * 32;
    v8f s[2][2] = {{(v8f){}, (v8f){}}, {(v8f){}, (v8f){}}};
    #pragma unroll
    for (int ks = 0; ks < 2; ++ks) {
      v16bf b0 = load_b_frag(kp, k0,      kDh, 32 * ks);
      v16bf b1 = load_b_frag(kp, k0 + 16, kDh, 32 * ks);
      s[0][0] = wmma_bf16(qa[0][ks], b0, s[0][0]);
      s[0][1] = wmma_bf16(qa[0][ks], b1, s[0][1]);
      s[1][0] = wmma_bf16(qa[1][ks], b0, s[1][0]);
      s[1][1] = wmma_bf16(qa[1][ks], b1, s[1][1]);
    }
    // scale + causal mask + online softmax
    #pragma unroll
    for (int mf = 0; mf < 2; ++mf)
      #pragma unroll
      for (int r = 0; r < 8; ++r) {
        int qrow = q0 + 16 * mf + 8 * hh + r;
        #pragma unroll
        for (int nf = 0; nf < 2; ++nf) {
          int key = k0 + 16 * nf + colL;
          float sv = s[mf][nf][r] * 0.125f;     // 1/sqrt(64)
          s[mf][nf][r] = (key <= qrow) ? sv : -1.0e30f;
        }
        float tmax = fmaxf(s[mf][0][r], s[mf][1][r]);
        #pragma unroll
        for (int off = 8; off >= 1; off >>= 1)
          tmax = fmaxf(tmax, __shfl_xor(tmax, off, 32));
        float mnew  = fmaxf(mrow[mf][r], tmax);
        float alpha = __expf(mrow[mf][r] - mnew);
        mrow[mf][r] = mnew;
        float p0 = __expf(s[mf][0][r] - mnew);
        float p1 = __expf(s[mf][1][r] - mnew);
        s[mf][0][r] = p0; s[mf][1][r] = p1;
        float ps = p0 + p1;
        #pragma unroll
        for (int off = 8; off >= 1; off >>= 1)
          ps += __shfl_xor(ps, off, 32);
        lrow[mf][r] = lrow[mf][r] * alpha + ps;
        #pragma unroll
        for (int nf = 0; nf < 4; ++nf) o[mf][nf][r] *= alpha;
      }
    // C-layout -> A-layout for P via per-wave LDS round trip (bf16)
    #pragma unroll
    for (int mf = 0; mf < 2; ++mf)
      #pragma unroll
      for (int nf = 0; nf < 2; ++nf)
        #pragma unroll
        for (int r = 0; r < 8; ++r)
          plds[wave][(16 * mf + 8 * hh + r) * 32 + 16 * nf + colL] = (__bf16)s[mf][nf][r];
    v16bf pa0 = load_a_frag(&plds[wave][0], 0,  32, 0);
    v16bf pa1 = load_a_frag(&plds[wave][0], 16, 32, 0);
    // O += P @ V  (V^T stored [Dh,T] row-major -> direct B fragments)
    #pragma unroll
    for (int nf = 0; nf < 4; ++nf) {
      v16bf vb = load_b_frag(vp, 16 * nf, kT, k0);
      o[0][nf] = wmma_bf16(pa0, vb, o[0][nf]);
      o[1][nf] = wmma_bf16(pa1, vb, o[1][nf]);
    }
  }
  // epilogue: normalize, write bf16 y_att in [B*T, C] layout (col = h*64+d)
  int b = bh >> 4, h = bh & 15;
  #pragma unroll
  for (int mf = 0; mf < 2; ++mf)
    #pragma unroll
    for (int nf = 0; nf < 4; ++nf)
      #pragma unroll
      for (int r = 0; r < 8; ++r) {
        int trow = q0 + 16 * mf + 8 * hh + r;
        int d = 16 * nf + colL;
        float val = o[mf][nf][r] / lrow[mf][r];
        yatt[((size_t)(b * kT + trow)) * kC + h * kDh + d] = (__bf16)val;
      }
}

// ---------------- output projection GEMM ----------------
__global__ void out_gemm(const __bf16* __restrict__ yb, const __bf16* __restrict__ wT,
                         const float* __restrict__ bias, float* __restrict__ y) {
  int wave = threadIdx.x >> 5;
  int m0 = blockIdx.x * 32;
  int n0 = (blockIdx.y * 4 + wave) * 32;
  v8f c00 = {}, c01 = {}, c10 = {}, c11 = {};
  for (int k0 = 0; k0 < kC; k0 += 32) {
    v16bf a0 = load_a_frag(yb, m0,      kC, k0);
    v16bf a1 = load_a_frag(yb, m0 + 16, kC, k0);
    v16bf b0 = load_b_frag(wT, n0,      kC, k0);
    v16bf b1 = load_b_frag(wT, n0 + 16, kC, k0);
    c00 = wmma_bf16(a0, b0, c00);
    c01 = wmma_bf16(a0, b1, c01);
    c10 = wmma_bf16(a1, b0, c10);
    c11 = wmma_bf16(a1, b1, c11);
  }
  int lane = threadIdx.x & 31;
  int colL = lane & 15, hh = lane >> 4;
  v8f acc[2][2] = {{c00, c01}, {c10, c11}};
  #pragma unroll
  for (int mf = 0; mf < 2; ++mf)
    #pragma unroll
    for (int nf = 0; nf < 2; ++nf)
      #pragma unroll
      for (int r = 0; r < 8; ++r) {
        int row = m0 + 16 * mf + 8 * hh + r;
        int n   = n0 + 16 * nf + colL;
        y[(size_t)row * kC + n] = acc[mf][nf][r] + bias[n];
      }
}

extern "C" void kernel_launch(void* const* d_in, const int* in_sizes, int n_in,
                              void* d_out, int out_size, void* d_ws, size_t ws_size,
                              hipStream_t stream) {
  (void)in_sizes; (void)n_in; (void)out_size; (void)ws_size;
  const float* x     = (const float*)d_in[0];
  const float* W_qkv = (const float*)d_in[1];
  const float* b_qkv = (const float*)d_in[2];
  const float* W_out = (const float*)d_in[3];
  const float* b_out = (const float*)d_in[4];

  float* y_out = (float*)d_out;                      // [2,2048,1024]
  float* k_out = y_out + (size_t)kBT * kC;           // [2,16,2048,64]
  float* v_out = k_out + (size_t)kBT * kC;           // [2,16,2048,64]

  char* ws = (char*)d_ws;
  __bf16* xb    = (__bf16*)(ws);
  __bf16* wqkvT = (__bf16*)(ws + OFF_WQKVT);
  __bf16* woutT = (__bf16*)(ws + OFF_WOUTT);
  __bf16* qb    = (__bf16*)(ws + OFF_QB);
  __bf16* kb    = (__bf16*)(ws + OFF_KB);
  __bf16* vTb   = (__bf16*)(ws + OFF_VTB);
  __bf16* yattb = (__bf16*)(ws + OFF_YATT);

  int nx = kBT * kC;
  cvt_f32_bf16<<<(nx + 255) / 256, 256, 0, stream>>>(x, xb, nx);
  cvt_transpose<<<(kC * kNq + 255) / 256, 256, 0, stream>>>(W_qkv, wqkvT, kC, kNq);
  cvt_transpose<<<(kC * kC + 255) / 256, 256, 0, stream>>>(W_out, woutT, kC, kC);

  dim3 g1(kBT / 32, kNq / 128);
  qkv_gemm<<<g1, 128, 0, stream>>>(xb, wqkvT, b_qkv, k_out, v_out, qb, kb, vTb);

  int attn_waves = kB * kH * (kT / 32);              // 2048
  attn_kernel<<<attn_waves / 4, 128, 0, stream>>>(qb, kb, vTb, yattb);

  dim3 g2(kBT / 32, kC / 128);
  out_gemm<<<g2, 128, 0, stream>>>(yattb, woutT, b_out, y_out);
}
